// GNNRegressor_70454643523891
// MI455X (gfx1250) — matrix operand order repaired and checked by reading, not compiled, not run on hardware
//
#include <hip/hip_runtime.h>
#include <hip/hip_bf16.h>
#include <math.h>

#define NN   50000
#define EE   800000
#define GG   512
#define HH   256
#define NF   32
#define EF   16
#define LL   4
#define BN_EPS 1e-5f

#define ROWS_PER_BLOCK 64

typedef _Float16 v8h  __attribute__((ext_vector_type(8)));
typedef _Float16 v16h __attribute__((ext_vector_type(16)));
typedef float    v8f  __attribute__((ext_vector_type(8)));

__device__ __forceinline__ v16h cat8(v8h a, v8h b) {
    return __builtin_shufflevector(a, b, 0,1,2,3,4,5,6,7,8,9,10,11,12,13,14,15);
}

// ---------------------------------------------------------------- zero
__global__ __launch_bounds__(256) void zero_kernel(float* __restrict__ p, int n) {
    int i = blockIdx.x * 256 + threadIdx.x;
    if (i < n) p[i] = 0.0f;
}

// ---------------------------------------------------------------- pack Wl1/Wl2 -> f16 B-fragment tiles
// Packed layout per 256x256 matrix (K x N): tile (kt,nt) of 32x16 at
// (nt*8+kt)*512 halves; within tile: lane = (k'>=16?16:0)+n', elem = k'&15.
__global__ __launch_bounds__(256) void pack_weights_kernel(
    const float* __restrict__ Wl1, const float* __restrict__ Wl2,
    _Float16* __restrict__ pk)
{
    int idx = blockIdx.x * 256 + threadIdx.x;      // 8 * 65536 total
    int mi  = idx >> 16;                           // matrix index 0..7
    int off = idx & 65535;
    int k = off >> 8, n = off & 255;
    int layer = mi >> 1;
    const float* src = (mi & 1) ? Wl2 : Wl1;
    float v = src[layer * HH * HH + k * HH + n];
    int kt = k >> 5, kk = k & 31;
    int nt = n >> 4, nn = n & 15;
    int lane = ((kk >> 4) << 4) + nn;
    int el   = kk & 15;
    pk[(size_t)mi * 65536 + (size_t)(nt * 8 + kt) * 512 + lane * 16 + el] = (_Float16)v;
}

// ---------------------------------------------------------------- h = x @ Win + b
__global__ __launch_bounds__(256) void input_gemm_kernel(
    const float* __restrict__ x, const float* __restrict__ Win,
    const float* __restrict__ b, float* __restrict__ h)
{
    __shared__ float sW[NF * HH];   // 32 KB
    __shared__ float sb[HH];
    int tid = threadIdx.x;
    for (int i = tid; i < NF * HH; i += 256) sW[i] = Win[i];
    sb[tid] = b[tid];
    __syncthreads();
    int n  = blockIdx.x * 4 + (tid >> 6);
    int c0 = (tid & 63) * 4;
    if (n >= NN) return;
    const float* xp = x + (size_t)n * NF;
    float a0 = sb[c0], a1 = sb[c0+1], a2 = sb[c0+2], a3 = sb[c0+3];
#pragma unroll
    for (int k = 0; k < NF; ++k) {
        float xv = xp[k];
        const float* w = &sW[k * HH + c0];
        a0 += xv * w[0]; a1 += xv * w[1]; a2 += xv * w[2]; a3 += xv * w[3];
    }
    float4 o = make_float4(a0, a1, a2, a3);
    *(float4*)&h[(size_t)n * HH + c0] = o;
}

// ---------------------------------------------------------------- fused edge: aggr += relu(h[src] + ea@We + be)
__global__ __launch_bounds__(256) void edge_kernel(
    const float* __restrict__ ea, const int* __restrict__ src,
    const int* __restrict__ dst,  const float* __restrict__ We,
    const float* __restrict__ be, const float* __restrict__ h,
    float* __restrict__ aggr)
{
    __shared__ float sW[EF * HH];   // 16 KB
    __shared__ float sb[HH];
    int tid = threadIdx.x;
    for (int i = tid; i < EF * HH; i += 256) sW[i] = We[i];
    sb[tid] = be[tid];
    __syncthreads();
    int e = blockIdx.x * 4 + (tid >> 6);
    if (e >= EE) return;
    int c0 = (tid & 63) * 4;
    int s = src[e], d = dst[e];
    float a0 = sb[c0], a1 = sb[c0+1], a2 = sb[c0+2], a3 = sb[c0+3];
    const float* ep = ea + (size_t)e * EF;
#pragma unroll
    for (int k = 0; k < EF; ++k) {
        float av = ep[k];
        const float* w = &sW[k * HH + c0];
        a0 += av * w[0]; a1 += av * w[1]; a2 += av * w[2]; a3 += av * w[3];
    }
    const float4 hv = *(const float4*)&h[(size_t)s * HH + c0];
    float m0 = fmaxf(hv.x + a0, 0.f), m1 = fmaxf(hv.y + a1, 0.f);
    float m2 = fmaxf(hv.z + a2, 0.f), m3 = fmaxf(hv.w + a3, 0.f);
    float* ap = aggr + (size_t)d * HH + c0;
    atomicAdd(ap + 0, m0); atomicAdd(ap + 1, m1);
    atomicAdd(ap + 2, m2); atomicAdd(ap + 3, m3);
}

// ---------------------------------------------------------------- WMMA node layer (in-place h update)
// block: 256 threads (8 waves), 64 rows. Each wave: 2 M-tiles x 4 N-tiles,
// so each B-fragment load feeds 2 WMMAs. t=h+aggr -> GEMM1(relu) -> GEMM2
// -> BN affine -> h += relu(.)
__global__ __launch_bounds__(256) void node_layer_kernel(
    float* __restrict__ hbuf, const float* __restrict__ aggr,
    const _Float16* __restrict__ W1p, const _Float16* __restrict__ W2p,
    const float* __restrict__ b1, const float* __restrict__ b2,
    const float* __restrict__ bnw, const float* __restrict__ bnb,
    const float* __restrict__ bnm, const float* __restrict__ bnv)
{
    __shared__ _Float16 ldsT[ROWS_PER_BLOCK * HH];   // 32 KB, GEMM1 input (f16)
    __shared__ _Float16 ldsU[ROWS_PER_BLOCK * HH];   // 32 KB, GEMM2 input (f16)

    int tid  = threadIdx.x;
    int wave = tid >> 5;         // 0..7
    int lane = tid & 31;
    int row0 = blockIdx.x * ROWS_PER_BLOCK;

    // stage t = h + aggr as f16 (64 rows x 256 cols = 4096 float4 chunks)
    for (int idx = tid; idx < ROWS_PER_BLOCK * 64; idx += 256) {
        int r  = idx >> 6;
        int c4 = (idx & 63) * 4;
        int o = r * HH + c4;
        int row = row0 + r;
        if (row < NN) {
            size_t go = (size_t)row * HH + c4;
            float4 hv = *(const float4*)&hbuf[go];
            float4 av = *(const float4*)&aggr[go];
            ldsT[o+0] = (_Float16)(hv.x + av.x);
            ldsT[o+1] = (_Float16)(hv.y + av.y);
            ldsT[o+2] = (_Float16)(hv.z + av.z);
            ldsT[o+3] = (_Float16)(hv.w + av.w);
        } else {
            ldsT[o+0] = (_Float16)0.f; ldsT[o+1] = (_Float16)0.f;
            ldsT[o+2] = (_Float16)0.f; ldsT[o+3] = (_Float16)0.f;
        }
    }
    __syncthreads();

    int ng = wave & 3;            // column group: cols [ng*64, ng*64+64)
    int mp = wave >> 2;           // M-pair: M-tiles {2*mp, 2*mp+1} (rows 32*mp..32*mp+31)
    int colbase = ng * 64;
    int arow  = lane & 15;
    int koff  = (lane >> 4) * 16;
    int mbase = (lane >> 4) * 8;
    int nloc  = lane & 15;

    // ---- GEMM1: u = relu(t @ W1 + b1)
    v8f acc[2][4] = {};
#pragma unroll
    for (int kt = 0; kt < 8; ++kt) {
        const _Float16* ap0 = &ldsT[((2*mp+0)*16 + arow) * HH + kt * 32 + koff];
        const _Float16* ap1 = &ldsT[((2*mp+1)*16 + arow) * HH + kt * 32 + koff];
        v16h a0 = cat8(*(const v8h*)ap0, *(const v8h*)(ap0 + 8));
        v16h a1 = cat8(*(const v8h*)ap1, *(const v8h*)(ap1 + 8));
#pragma unroll
        for (int nt = 0; nt < 4; ++nt) {
            const _Float16* bp = &W1p[(size_t)((ng * 4 + nt) * 8 + kt) * 512 + lane * 16];
            v16h bmat = *(const v16h*)bp;
            acc[0][nt] = __builtin_amdgcn_wmma_f32_16x16x32_f16(
                false, a0, false, bmat, (short)0, acc[0][nt], false, false);
            acc[1][nt] = __builtin_amdgcn_wmma_f32_16x16x32_f16(
                false, a1, false, bmat, (short)0, acc[1][nt], false, false);
        }
    }
#pragma unroll
    for (int nt = 0; nt < 4; ++nt) {
        int col  = colbase + nt * 16 + nloc;
        float bv = b1[col];
#pragma unroll
        for (int mi = 0; mi < 2; ++mi) {
            int mrow = (2*mp + mi) * 16 + mbase;
#pragma unroll
            for (int v = 0; v < 8; ++v) {
                float u = fmaxf(acc[mi][nt][v] + bv, 0.f);
                ldsU[(mrow + v) * HH + col] = (_Float16)u;
            }
        }
    }
    __syncthreads();

    // ---- GEMM2: t2 = u @ W2 + b2, then BN affine + relu + residual
    v8f acc2[2][4] = {};
#pragma unroll
    for (int kt = 0; kt < 8; ++kt) {
        const _Float16* ap0 = &ldsU[((2*mp+0)*16 + arow) * HH + kt * 32 + koff];
        const _Float16* ap1 = &ldsU[((2*mp+1)*16 + arow) * HH + kt * 32 + koff];
        v16h a0 = cat8(*(const v8h*)ap0, *(const v8h*)(ap0 + 8));
        v16h a1 = cat8(*(const v8h*)ap1, *(const v8h*)(ap1 + 8));
#pragma unroll
        for (int nt = 0; nt < 4; ++nt) {
            const _Float16* bp = &W2p[(size_t)((ng * 4 + nt) * 8 + kt) * 512 + lane * 16];
            v16h bmat = *(const v16h*)bp;
            acc2[0][nt] = __builtin_amdgcn_wmma_f32_16x16x32_f16(
                false, a0, false, bmat, (short)0, acc2[0][nt], false, false);
            acc2[1][nt] = __builtin_amdgcn_wmma_f32_16x16x32_f16(
                false, a1, false, bmat, (short)0, acc2[1][nt], false, false);
        }
    }
#pragma unroll
    for (int nt = 0; nt < 4; ++nt) {
        int col = colbase + nt * 16 + nloc;
        float bv    = b2[col];
        float scale = bnw[col] * rsqrtf(bnv[col] + BN_EPS);
        float shift = bnb[col] - bnm[col] * scale;
#pragma unroll
        for (int mi = 0; mi < 2; ++mi) {
            int mrow = (2*mp + mi) * 16 + mbase;
#pragma unroll
            for (int v = 0; v < 8; ++v) {
                int row = row0 + mrow + v;
                if (row < NN) {
                    size_t go = (size_t)row * HH + col;
                    float t2 = acc2[mi][nt][v] + bv;
                    float tn = t2 * scale + shift;
                    hbuf[go] = hbuf[go] + fmaxf(tn, 0.f);
                }
            }
        }
    }
}

// ---------------------------------------------------------------- g = segment_sum(h, batch)
__global__ __launch_bounds__(256) void pool_kernel(
    const float* __restrict__ h, const int* __restrict__ batch,
    float* __restrict__ g)
{
    int idx = blockIdx.x * 256 + threadIdx.x;   // N * 64
    int n  = idx >> 6;
    int c0 = (idx & 63) * 4;
    if (n >= NN) return;
    int b = batch[n];
    float4 v = *(const float4*)&h[(size_t)n * HH + c0];
    float* gp = g + (size_t)b * HH + c0;
    atomicAdd(gp + 0, v.x); atomicAdd(gp + 1, v.y);
    atomicAdd(gp + 2, v.z); atomicAdd(gp + 3, v.w);
}

// ---------------------------------------------------------------- per-graph heads (z, edl)
__global__ __launch_bounds__(128) void head_kernel(
    const float* __restrict__ g,   const float* __restrict__ gfeat,
    const float* __restrict__ Wp1, const float* __restrict__ bp1,
    const float* __restrict__ Wp2, const float* __restrict__ bp2,
    const float* __restrict__ Wf1, const float* __restrict__ bf1,
    const float* __restrict__ Wf2, const float* __restrict__ bf2,
    const float* __restrict__ Wh,  const float* __restrict__ bh,
    float* __restrict__ edl, float* __restrict__ z)
{
    __shared__ float sg[HH + NF];   // gc = [g, global_feat]
    __shared__ float a1[HH];
    __shared__ float a2[HH / 2];
    __shared__ float p1[HH / 2];
    __shared__ float zz[64];
    __shared__ float red[128];
    int b = blockIdx.x, tid = threadIdx.x;

    const float* grow = g + (size_t)b * HH;
    for (int i = tid; i < HH; i += 128) sg[i] = grow[i];
    if (tid < NF) sg[HH + tid] = gfeat[(size_t)b * NF + tid];
    __syncthreads();

    // z path stage 1: p1 = relu(g @ Wp1 + bp1)   (128 outputs)
    {
        float acc = bp1[tid];
        for (int k = 0; k < HH; ++k) acc += sg[k] * Wp1[k * 128 + tid];
        p1[tid] = fmaxf(acc, 0.f);
    }
    // f path stage 1: a1 = relu(gc @ Wf1 + bf1)  (256 outputs)
    for (int j = tid; j < HH; j += 128) {
        float acc = bf1[j];
        for (int k = 0; k < HH + NF; ++k) acc += sg[k] * Wf1[k * HH + j];
        a1[j] = fmaxf(acc, 0.f);
    }
    __syncthreads();

    // z path stage 2: zz = p1 @ Wp2 + bp2        (64 outputs)
    if (tid < 64) {
        float acc = bp2[tid];
        for (int k = 0; k < 128; ++k) acc += p1[k] * Wp2[k * 64 + tid];
        zz[tid] = acc;
    }
    // f path stage 2: a2 = relu(a1 @ Wf2 + bf2)  (128 outputs)
    {
        float acc = bf2[tid];
        for (int k = 0; k < HH; ++k) acc += a1[k] * Wf2[k * 128 + tid];
        a2[tid] = fmaxf(acc, 0.f);
    }
    __syncthreads();

    // ||zz|| reduction
    red[tid] = (tid < 64) ? zz[tid] * zz[tid] : 0.f;
    __syncthreads();
    for (int w = 64; w > 0; w >>= 1) {
        if (tid < w) red[tid] += red[tid + w];
        __syncthreads();
    }
    float inv = 1.f / fmaxf(sqrtf(red[0]), 1e-12f);
    if (tid < 64) z[(size_t)b * 64 + tid] = zz[tid] * inv;

    // EDL head: o = a2 @ Whead + bhead (4 outputs)
    if (tid < 4) {
        float acc = bh[tid];
        for (int k = 0; k < 128; ++k) acc += a2[k] * Wh[k * 4 + tid];
        float r;
        if (tid == 0) {
            r = acc;
        } else {
            float sp = (acc > 20.f) ? acc : log1pf(expf(acc));
            r = sp + ((tid == 2) ? (1.f + 1e-6f) : 1e-6f);
        }
        edl[(size_t)b * 4 + tid] = r;
    }
}

// ----------------------------------------------------------------
extern "C" void kernel_launch(void* const* d_in, const int* in_sizes, int n_in,
                              void* d_out, int out_size, void* d_ws, size_t ws_size,
                              hipStream_t stream) {
    const float* x      = (const float*)d_in[0];
    const int*   ei     = (const int*)  d_in[1];
    const float* ea     = (const float*)d_in[2];
    const int*   batch  = (const int*)  d_in[3];
    const float* gfeat  = (const float*)d_in[4];
    const float* Win    = (const float*)d_in[5];
    const float* bin_   = (const float*)d_in[6];
    const float* We     = (const float*)d_in[7];
    const float* be     = (const float*)d_in[8];
    const float* Wl1    = (const float*)d_in[9];
    const float* bl1    = (const float*)d_in[10];
    const float* Wl2    = (const float*)d_in[11];
    const float* bl2    = (const float*)d_in[12];
    const float* bnw    = (const float*)d_in[13];
    const float* bnb    = (const float*)d_in[14];
    const float* bnm    = (const float*)d_in[15];
    const float* bnv    = (const float*)d_in[16];
    const float* Wp1    = (const float*)d_in[17];
    const float* bp1    = (const float*)d_in[18];
    const float* Wp2    = (const float*)d_in[19];
    const float* bp2    = (const float*)d_in[20];
    const float* Wf1    = (const float*)d_in[21];
    const float* bf1    = (const float*)d_in[22];
    const float* Wf2    = (const float*)d_in[23];
    const float* bf2    = (const float*)d_in[24];
    const float* Whead  = (const float*)d_in[25];
    const float* bhead  = (const float*)d_in[26];
    (void)in_sizes; (void)n_in; (void)out_size; (void)ws_size;

    const int* src = ei;          // edge_index[0]
    const int* dst = ei + EE;     // edge_index[1]

    // workspace layout
    float*    h    = (float*)d_ws;                       // N*H
    float*    aggr = h + (size_t)NN * HH;                // N*H
    _Float16* wpk  = (_Float16*)(aggr + (size_t)NN * HH);// 8 * 65536 halves

    // d_out layout: edl[512*4] | z[512*64] | g[512*256]
    float* edl = (float*)d_out;
    float* z   = edl + (size_t)GG * 4;
    float* g   = z   + (size_t)GG * 64;

    // 1. pack Wl1/Wl2 into f16 WMMA B-fragment tiles
    pack_weights_kernel<<<(8 * 65536) / 256, 256, 0, stream>>>(Wl1, Wl2, wpk);

    // 2. h = x @ Win + bin
    input_gemm_kernel<<<NN / 4, 256, 0, stream>>>(x, Win, bin_, h);

    // 3. message-passing layers
    for (int i = 0; i < LL; ++i) {
        zero_kernel<<<(NN * HH + 255) / 256, 256, 0, stream>>>(aggr, NN * HH);
        edge_kernel<<<EE / 4, 256, 0, stream>>>(
            ea, src, dst, We + (size_t)i * EF * HH, be + (size_t)i * HH, h, aggr);
        node_layer_kernel<<<(NN + ROWS_PER_BLOCK - 1) / ROWS_PER_BLOCK, 256, 0, stream>>>(
            h, aggr,
            wpk + (size_t)(2 * i) * 65536, wpk + (size_t)(2 * i + 1) * 65536,
            bl1 + (size_t)i * HH, bl2 + (size_t)i * HH,
            bnw + (size_t)i * HH, bnb + (size_t)i * HH,
            bnm + (size_t)i * HH, bnv + (size_t)i * HH);
    }

    // 4. graph pooling into g (lives in d_out)
    zero_kernel<<<(GG * HH + 255) / 256, 256, 0, stream>>>(g, GG * HH);
    pool_kernel<<<(NN * 64 + 255) / 256, 256, 0, stream>>>(h, batch, g);

    // 5. heads
    head_kernel<<<GG, 128, 0, stream>>>(g, gfeat, Wp1, bp1, Wp2, bp2,
                                        Wf1, bf1, Wf2, bf2, Whead, bhead, edl, z);
}
